// LSTM_25709674233938
// MI455X (gfx1250) — matrix operand order repaired and checked by reading, not compiled, not run on hardware
//
#include <hip/hip_runtime.h>
#include <hip/hip_bf16.h>

// ---------------------------------------------------------------------------
// LSTM (T=512, B=256, D_IN=85, H=512, N_OUT=33) for gfx1250 (MI455X, wave32).
//
//  Kernel 0: reset barrier counter.
//  Kernel 1: convert x (f32) -> bf16 padded to K=96, 16B-aligned rows (ws).
//  Kernel 2 (persistent, 64 WGs): time scan. Recurrent + input weights pinned
//           in LDS as bf16 WMMA B-fragments; cell state in registers; global
//           atomic barrier between timesteps; hidden states stored bf16 to ws.
//  Kernel 3: readout GEMM [T*B,512] x [512,33->48] with WMMA.
//
//  Workspace: [0..4)                  barrier counter
//             [256 .. +128MB)         hseq bf16 [T][B][H]
//             [256+128MB .. +24MB)    xbf  bf16 [T*B][96]
// ---------------------------------------------------------------------------

typedef __attribute__((ext_vector_type(16))) __bf16 v16bf;
typedef __attribute__((ext_vector_type(8)))  float  v8f;

#define T_STEPS 512
#define BATCH   256
#define DIN     85
#define DINP    96               // DIN padded to multiple of 32
#define HDIM    512
#define N_OUT   33

#define HC 16                    // h-columns per workgroup
#define BC 128                   // batch rows per workgroup (8 waves x 16)
#define NHCH (HDIM / HC)         // 32 h-chunks
#define NBCH (BATCH / BC)        // 2 batch-chunks
#define NWG  (NHCH * NBCH)       // 64 persistent workgroups
#define KH_STEPS (HDIM / 32)     // 16 K-steps for hidden GEMM
#define KX_STEPS (DINP / 32)     // 3 K-steps for input GEMM

__device__ __forceinline__ unsigned short f2bf(float f) {
  unsigned u = __float_as_uint(f);
  u += 0x7FFFu + ((u >> 16) & 1u);     // round-to-nearest-even
  return (unsigned short)(u >> 16);
}

union FragBF { v16bf v; unsigned short h[16]; uint4 q[2]; };

__device__ __forceinline__ float sigmoidf_(float z) {
  return 1.0f / (1.0f + __expf(-z));
}

__global__ void lstm_init_ctr(unsigned* ctr) {
  if (threadIdx.x == 0) *ctr = 0u;
}

// ---------------------------------------------------------------------------
// x f32 [T*B][85]  ->  xbf bf16 [T*B][96] (zero-padded), one uint4 per thread
// ---------------------------------------------------------------------------
__global__ void __launch_bounds__(256)
lstm_xcvt(const float* __restrict__ x, unsigned short* __restrict__ xbf)
{
  int idx = blockIdx.x * 256 + threadIdx.x;          // one 8-element chunk
  const int total = T_STEPS * BATCH * (DINP / 8);    // 1,572,864
  if (idx >= total) return;
  int row = idx / (DINP / 8);
  int c   = idx % (DINP / 8);
  const float* xr = x + (size_t)row * DIN;
  union { unsigned short h[8]; uint4 q; } o;
#pragma unroll
  for (int e = 0; e < 8; ++e) {
    int k = c * 8 + e;
    o.h[e] = f2bf((k < DIN) ? xr[k] : 0.0f);
  }
  *(uint4*)(xbf + (size_t)row * DINP + c * 8) = o.q;
}

// ---------------------------------------------------------------------------
// Persistent scan kernel: grid = 64 WGs, 256 threads (8 waves) each.
// Wave w of WG (bchunk,hchunk) owns output tile rows [b0+16w, b0+16w+16),
// cols [h0, h0+16), all 4 gates; C-state persists in its registers.
// ---------------------------------------------------------------------------
__global__ void __launch_bounds__(256, 1)
lstm_scan(const unsigned short* __restrict__ xbf,
          const float* __restrict__ Wxi, const float* __restrict__ Wxf,
          const float* __restrict__ Wxo, const float* __restrict__ Wxc,
          const float* __restrict__ Whi, const float* __restrict__ Whf,
          const float* __restrict__ Who, const float* __restrict__ Whc,
          const float* __restrict__ bhi, const float* __restrict__ bhf,
          const float* __restrict__ bho, const float* __restrict__ bhc,
          unsigned* __restrict__ ctr,
          unsigned short* __restrict__ hseq)
{
  extern __shared__ char smem[];
  unsigned short* WhS = (unsigned short*)smem;                       // [4][16][32 lanes][16] bf16 = 64KB
  unsigned short* WxS = (unsigned short*)(smem + 4*KH_STEPS*512*2);  // [4][3][32][16]  = 12KB
  float*        biasS = (float*)(smem + 4*KH_STEPS*512*2 + 4*KX_STEPS*512*2); // [4][16]

  const int tid  = threadIdx.x;
  const int wave = tid >> 5;
  const int lane = tid & 31;
  const int half = lane >> 4;
  const int n    = lane & 15;      // A-row-in-tile / C-column

  const int hchunk = blockIdx.x & (NHCH - 1);
  const int bchunk = blockIdx.x >> 5;          // NHCH == 32
  const int h0   = hchunk * HC;
  const int b0   = bchunk * BC;
  const int rowA = b0 + wave * 16 + n;         // batch row this lane loads for A

  // ---- one-time: pre-swizzle weights into WMMA B-fragment layout in LDS ----
  for (int idx = tid; idx < 4 * KH_STEPS * 512; idx += 256) {
    int g  = idx >> 13;            // / 8192
    int r  = idx & 8191;
    int ks = r >> 9;
    int li = r & 511;
    int ln = li >> 4;
    int e  = li & 15;
    int col = h0 + (ln & 15);                  // output h index (B-matrix N)
    int k   = ks * 32 + (ln >> 4) * 8 + (e & 7) + (e >> 3) * 16;    // reduction index (B-matrix K)
    const float* W = (g == 0) ? Whi : (g == 1) ? Whf : (g == 2) ? Who : Whc;
    WhS[idx] = f2bf(W[col * HDIM + k]);
  }
  for (int idx = tid; idx < 4 * KX_STEPS * 512; idx += 256) {
    int g  = idx / (KX_STEPS * 512);
    int r  = idx % (KX_STEPS * 512);
    int ks = r >> 9;
    int li = r & 511;
    int ln = li >> 4;
    int e  = li & 15;
    int col = h0 + (ln & 15);
    int k   = ks * 32 + (ln >> 4) * 8 + (e & 7) + (e >> 3) * 16;
    const float* W = (g == 0) ? Wxi : (g == 1) ? Wxf : (g == 2) ? Wxo : Wxc;
    WxS[idx] = f2bf((k < DIN) ? W[col * DIN + k] : 0.0f);
  }
  if (tid < 64) {
    int g = tid >> 4, nn = tid & 15;
    const float* bb = (g == 0) ? bhi : (g == 1) ? bhf : (g == 2) ? bho : bhc;
    biasS[tid] = bb[h0 + nn];
  }
  __syncthreads();

  v8f Cst = {0, 0, 0, 0, 0, 0, 0, 0};          // persistent cell state tile

  for (int t = 0; t < T_STEPS; ++t) {
    v8f acc[4];
#pragma unroll
    for (int g = 0; g < 4; ++g) acc[g] = (v8f){0, 0, 0, 0, 0, 0, 0, 0};

    // prefetch next timestep's x fragment row (global_prefetch_b8)
    if (t + 1 < T_STEPS)
      __builtin_prefetch((const char*)(xbf + ((size_t)(t + 1) * BATCH + rowA) * DINP), 0, 1);

    // ---- input projection: pre-converted bf16 x, K = 96 ----
    const unsigned short* xrow = xbf + ((size_t)t * BATCH + rowA) * DINP;
#pragma unroll
    for (int ks = 0; ks < KX_STEPS; ++ks) {
      FragBF a;
      int kb = ks * 32 + half * 8;
      a.q[0] = *(const uint4*)(xrow + kb);
      a.q[1] = *(const uint4*)(xrow + kb + 16);
#pragma unroll
      for (int g = 0; g < 4; ++g) {
        FragBF b;
        const uint4* p = (const uint4*)(WxS + ((g * KX_STEPS + ks) * 32 + lane) * 16);
        b.q[0] = p[0]; b.q[1] = p[1];
        acc[g] = __builtin_amdgcn_wmma_f32_16x16x32_bf16(
            false, a.v, false, b.v, (short)0, acc[g], false, false);
      }
    }

    // ---- recurrent projection: H_{t-1} (bf16 from ws, L2-resident) --------
    if (t > 0) {
      const unsigned short* hrow = hseq + ((size_t)(t - 1) * BATCH + rowA) * HDIM;
      for (int ks = 0; ks < KH_STEPS; ++ks) {
        FragBF a;
        int kb = ks * 32 + half * 8;
        a.q[0] = *(const uint4*)(hrow + kb);
        a.q[1] = *(const uint4*)(hrow + kb + 16);
#pragma unroll
        for (int g = 0; g < 4; ++g) {
          FragBF b;
          const uint4* p = (const uint4*)(WhS + ((g * KH_STEPS + ks) * 32 + lane) * 16);
          b.q[0] = p[0]; b.q[1] = p[1];
          acc[g] = __builtin_amdgcn_wmma_f32_16x16x32_bf16(
              false, a.v, false, b.v, (short)0, acc[g], false, false);
        }
      }
    }

    // ---- gates (fp32 VALU), cell update in registers, H_t -> ws (bf16) ----
    float bi = biasS[n], bf_ = biasS[16 + n], bo_ = biasS[32 + n], bc_ = biasS[48 + n];
    unsigned short* hout = hseq + (size_t)t * BATCH * HDIM;
#pragma unroll
    for (int v = 0; v < 8; ++v) {
      float I  = sigmoidf_(acc[0][v] + bi);
      float F  = sigmoidf_(acc[1][v] + bf_);
      float O  = sigmoidf_(acc[2][v] + bo_);
      float Ch = tanhf(acc[3][v] + bc_);
      float C  = F * Cst[v] + I * Ch;
      Cst[v]   = C;
      float Hv = O * tanhf(C);
      int m    = v + half * 8;                 // C/D layout: row = v + 8*(lane>=16)
      int row  = b0 + wave * 16 + m;
      hout[(size_t)row * HDIM + h0 + n] = f2bf(Hv);
    }

    // ---- grid-wide step barrier (monotonic counter) ----
    __threadfence();
    __syncthreads();
    if (tid == 0) {
      atomicAdd(ctr, 1u);
      unsigned target = (unsigned)(t + 1) * NWG;
      while (*(volatile unsigned*)ctr < target) __builtin_amdgcn_s_sleep(1);
    }
    __syncthreads();
    __threadfence();
  }
}

// ---------------------------------------------------------------------------
// Readout: out[t,b,o] = hseq[t,b,:] . Wro[o,:] + bro[o].  M = T*B = 131072,
// N = 33 padded to 48 (3 col-tiles), K = 512. Wave -> one 16-row slab.
// ---------------------------------------------------------------------------
__global__ void __launch_bounds__(256, 2)
lstm_readout(const unsigned short* __restrict__ hseq,
             const float* __restrict__ Wro, const float* __restrict__ bro,
             float* __restrict__ out)
{
  extern __shared__ char smem[];
  unsigned short* WoS = (unsigned short*)smem;   // [3][16][32][16] bf16 = 48KB

  const int tid  = threadIdx.x;
  const int wave = tid >> 5;
  const int lane = tid & 31;
  const int half = lane >> 4;
  const int n    = lane & 15;

  for (int idx = tid; idx < 3 * KH_STEPS * 512; idx += 256) {
    int ct = idx >> 13;
    int r  = idx & 8191;
    int ks = r >> 9;
    int li = r & 511;
    int ln = li >> 4;
    int e  = li & 15;
    int o  = ct * 16 + (ln & 15);
    int k  = ks * 32 + (ln >> 4) * 8 + (e & 7) + (e >> 3) * 16;
    WoS[idx] = f2bf((o < N_OUT) ? Wro[o * HDIM + k] : 0.0f);
  }
  __syncthreads();

  const int rt   = blockIdx.x * 8 + wave;        // 16-row tile of T*B
  const int rowA = rt * 16 + n;
  const unsigned short* hrow = hseq + (size_t)rowA * HDIM;

  v8f acc[3];
#pragma unroll
  for (int ct = 0; ct < 3; ++ct) acc[ct] = (v8f){0, 0, 0, 0, 0, 0, 0, 0};

  for (int ks = 0; ks < KH_STEPS; ++ks) {
    FragBF a;
    int kb = ks * 32 + half * 8;
    a.q[0] = *(const uint4*)(hrow + kb);
    a.q[1] = *(const uint4*)(hrow + kb + 16);
#pragma unroll
    for (int ct = 0; ct < 3; ++ct) {
      FragBF b;
      const uint4* p = (const uint4*)(WoS + ((ct * KH_STEPS + ks) * 32 + lane) * 16);
      b.q[0] = p[0]; b.q[1] = p[1];
      acc[ct] = __builtin_amdgcn_wmma_f32_16x16x32_bf16(
          false, a.v, false, b.v, (short)0, acc[ct], false, false);
    }
  }

#pragma unroll
  for (int ct = 0; ct < 3; ++ct) {
    int o = ct * 16 + n;
    if (o < N_OUT) {
      float bv = bro[o];
#pragma unroll
      for (int v = 0; v < 8; ++v) {
        int m = v + half * 8;
        size_t row = (size_t)rt * 16 + m;
        out[row * N_OUT + o] = acc[ct][v] + bv;
      }
    }
  }
}

// ---------------------------------------------------------------------------
extern "C" void kernel_launch(void* const* d_in, const int* in_sizes, int n_in,
                              void* d_out, int out_size, void* d_ws, size_t ws_size,
                              hipStream_t stream) {
  (void)in_sizes; (void)n_in; (void)out_size; (void)ws_size;
  const float* x   = (const float*)d_in[0];
  const float* Wxi = (const float*)d_in[1];
  const float* Wxf = (const float*)d_in[2];
  const float* Wxo = (const float*)d_in[3];
  const float* Wxc = (const float*)d_in[4];
  const float* Whi = (const float*)d_in[5];
  const float* bhi = (const float*)d_in[6];
  const float* Whf = (const float*)d_in[7];
  const float* bhf = (const float*)d_in[8];
  const float* Who = (const float*)d_in[9];
  const float* bho = (const float*)d_in[10];
  const float* Whc = (const float*)d_in[11];
  const float* bhc = (const float*)d_in[12];
  const float* Wro = (const float*)d_in[13];
  const float* bro = (const float*)d_in[14];
  float* out = (float*)d_out;

  unsigned* ctr = (unsigned*)d_ws;
  unsigned short* hseq = (unsigned short*)((char*)d_ws + 256);           // 128MB bf16 [T][B][H]
  unsigned short* xbf  = (unsigned short*)((char*)d_ws + 256 +
                          (size_t)T_STEPS * BATCH * HDIM * 2);           // 24MB bf16 [T*B][96]

  lstm_init_ctr<<<1, 64, 0, stream>>>(ctr);

  int xcvt_chunks = T_STEPS * BATCH * (DINP / 8);
  lstm_xcvt<<<dim3((xcvt_chunks + 255) / 256), dim3(256), 0, stream>>>(x, xbf);

  size_t shScan = (size_t)(4 * KH_STEPS * 512 * 2 + 4 * KX_STEPS * 512 * 2 + 64 * 4); // 78080B
  lstm_scan<<<dim3(NWG), dim3(256), shScan, stream>>>(
      xbf, Wxi, Wxf, Wxo, Wxc, Whi, Whf, Who, Whc,
      bhi, bhf, bho, bhc, ctr, hseq);

  size_t shRead = (size_t)(3 * KH_STEPS * 512 * 2);  // 49152B
  lstm_readout<<<dim3((T_STEPS * BATCH / 16) / 8), dim3(256), shRead, stream>>>(
      hseq, Wro, bro, out);
}
